// GCN_28432683499972
// MI455X (gfx1250) — matrix operand order repaired and checked by reading, not compiled, not run on hardware
//
#include <hip/hip_runtime.h>

#define SLOPE 0.01f

static const int kN = 50000;      // nodes (divisible by 16)
static const int kE = 800000;     // edges
static const int kLM = 768;       // lm input dim
static const int kH  = 192;       // hidden dim
static const int kTiles = kN / 16;        // 3125 row tiles
static const int kWavesPerBlock = 8;      // 256 threads

typedef __attribute__((ext_vector_type(16))) _Float16 v16h;
typedef __attribute__((ext_vector_type(8)))  _Float16 v8h;
typedef __attribute__((ext_vector_type(8)))  float    v8f;

__device__ __forceinline__ float lrelu(float v) { return v > 0.f ? v : v * SLOPE; }

__device__ __forceinline__ v8f wmma_f16(v16h a, v16h b, v8f c) {
    return __builtin_amdgcn_wmma_f32_16x16x32_f16(
        /*neg_a=*/false, a, /*neg_b=*/false, b,
        /*c_mod=*/(short)0, c, /*reuse_a=*/false, /*reuse_b=*/false);
}

// ---- fragment loaders (CDNA5 wave32 WMMA layouts) ----
// A 16x32 f16 fragment: lane L holds row M=L&15.
//   lanes 0-15 : K = k0+0..7  (elems 0-7), K = k0+16..23 (elems 8-15)
//   lanes 16-31: K = k0+8..15 (elems 0-7), K = k0+24..31 (elems 8-15)
__device__ __forceinline__ v16h load_a_f16(const _Float16* __restrict__ A, int ldk,
                                           int r0, int k0, int lane) {
    int m = lane & 15, kh = (lane >> 4) << 3;
    const _Float16* p = A + (size_t)(r0 + m) * ldk + k0 + kh;
    v8h lo = *(const v8h*)(p);
    v8h hi = *(const v8h*)(p + 16);
    return __builtin_shufflevector(lo, hi, 0,1,2,3,4,5,6,7,8,9,10,11,12,13,14,15);
}

__device__ __forceinline__ v16h load_a_f32(const float* __restrict__ A, int ldk,
                                           int r0, int k0, int lane) {
    int m = lane & 15, kh = (lane >> 4) << 3;
    const float* p = A + (size_t)(r0 + m) * ldk + k0 + kh;
    v16h a;
#pragma unroll
    for (int e = 0; e < 8; ++e) a[e] = (_Float16)p[e];
#pragma unroll
    for (int e = 0; e < 8; ++e) a[8 + e] = (_Float16)p[16 + e];
    return a;
}

// B 32x16 f16 fragment from f16 LDS copy of torch-layout weights W[Mout, K]
// (B[k][n] = W[n][k]). lane L holds col N=L&15;
// lanes 0-15: K=k0..7, k0+16..23; lanes 16-31: K=k0+8..15, k0+24..31
__device__ __forceinline__ v16h load_b_lds(const _Float16* Wl, int stride,
                                           int col0, int k0, int lane) {
    int n = lane & 15;
    int kb = k0 + ((lane >> 4) << 3);
    const _Float16* p = Wl + (col0 + n) * stride + kb;
    v8h lo = *(const v8h*)(p);
    v8h hi = *(const v8h*)(p + 16);
    return __builtin_shufflevector(lo, hi, 0,1,2,3,4,5,6,7,8,9,10,11,12,13,14,15);
}

// ---- tiny branches: num_prop (K=5) -> cols 0..31, num_category (K=1) -> cols 32..63
__global__ __launch_bounds__(256)
void small_branches(const float* __restrict__ prop, const float* __restrict__ cat,
                    const float* __restrict__ Wp, const float* __restrict__ bp,
                    const float* __restrict__ Wc, const float* __restrict__ bc,
                    _Float16* __restrict__ H) {
    int i = blockIdx.x * blockDim.x + threadIdx.x;
    if (i >= kN) return;
    float p0 = prop[i*5+0], p1 = prop[i*5+1], p2 = prop[i*5+2],
          p3 = prop[i*5+3], p4 = prop[i*5+4];
    float c0 = cat[i];
    _Float16* hrow = H + (size_t)i * kH;
#pragma unroll 4
    for (int j = 0; j < 32; ++j) {
        float v = bp[j] + p0*Wp[j*5+0] + p1*Wp[j*5+1] + p2*Wp[j*5+2]
                        + p3*Wp[j*5+3] + p4*Wp[j*5+4];
        hrow[j] = (_Float16)lrelu(v);
        float v2 = bc[j] + c0 * Wc[j];
        hrow[32 + j] = (_Float16)lrelu(v2);
    }
}

// ---- branch GEMM: [N,768] f32 @ W[32,768]^T + b, lrelu, write f16 into H cols colOff..+31
// Weights staged once per block in LDS as f16 (padded rows: 768 -> 776).
__global__ __launch_bounds__(256)
void branch_gemm768(const float* __restrict__ A, const float* __restrict__ W,
                    const float* __restrict__ bias, _Float16* __restrict__ H, int colOff) {
    constexpr int WSTR = 776;                 // padded f16 row stride
    __shared__ _Float16 Wl[32 * WSTR];
    for (int idx = threadIdx.x; idx < 32 * kLM; idx += 256) {
        int r = idx / kLM, c = idx - r * kLM;
        Wl[r * WSTR + c] = (_Float16)W[idx];
    }
    __syncthreads();

    int wave = blockIdx.x * (blockDim.x >> 5) + (threadIdx.x >> 5);
    int lane = threadIdx.x & 31;
    if (wave >= kTiles) return;
    int r0 = wave * 16;
    v8f c0, c1;
#pragma unroll
    for (int i = 0; i < 8; ++i) { c0[i] = 0.f; c1[i] = 0.f; }
    for (int k0 = 0; k0 < kLM; k0 += 32) {
        v16h a  = load_a_f32(A, kLM, r0, k0, lane);
        v16h b0 = load_b_lds(Wl, WSTR, 0,  k0, lane);
        v16h b1 = load_b_lds(Wl, WSTR, 16, k0, lane);
        c0 = wmma_f16(a, b0, c0);
        c1 = wmma_f16(a, b1, c1);
    }
    int nn = lane & 15, mb = (lane >> 4) << 3;
    float bv0 = bias[nn], bv1 = bias[16 + nn];
#pragma unroll
    for (int r = 0; r < 8; ++r) {
        size_t row = (size_t)(r0 + mb + r);
        H[row * kH + colOff + nn]      = (_Float16)lrelu(c0[r] + bv0);
        H[row * kH + colOff + 16 + nn] = (_Float16)lrelu(c1[r] + bv1);
    }
}

// ---- generic hidden GEMM: [N,192] f16 @ W[MOUT,192]^T, W staged in LDS as f16
// MODE 0: +bias, lrelu, store f16   (W_in)
// MODE 1: raw, store f32           (W_c1 / W_c2 -> hW)
// MODE 2: +bias, lrelu, store f32  (W_o1 -> em)
template <int MOUT, int MODE>
__global__ __launch_bounds__(256)
void gemm_h(const _Float16* __restrict__ A, const float* __restrict__ W,
            const float* __restrict__ bias, _Float16* __restrict__ outH,
            float* __restrict__ outF) {
    constexpr int WSTR = 200;                 // padded f16 row stride (192 -> 200)
    __shared__ _Float16 Wl[MOUT * WSTR];
    for (int idx = threadIdx.x; idx < MOUT * kH; idx += 256) {
        int r = idx / kH, c = idx - r * kH;
        Wl[r * WSTR + c] = (_Float16)W[idx];
    }
    __syncthreads();

    int wave = blockIdx.x * (blockDim.x >> 5) + (threadIdx.x >> 5);
    int lane = threadIdx.x & 31;
    if (wave >= kTiles) return;
    int r0 = wave * 16;
    constexpr int NT = MOUT / 16;
    v8f acc[NT];
#pragma unroll
    for (int t = 0; t < NT; ++t)
#pragma unroll
        for (int i = 0; i < 8; ++i) acc[t][i] = 0.f;
#pragma unroll
    for (int k0 = 0; k0 < kH; k0 += 32) {
        v16h a = load_a_f16(A, kH, r0, k0, lane);
#pragma unroll
        for (int t = 0; t < NT; ++t) {
            v16h b = load_b_lds(Wl, WSTR, t * 16, k0, lane);
            acc[t] = wmma_f16(a, b, acc[t]);
        }
    }
    int nn = lane & 15, mb = (lane >> 4) << 3;
#pragma unroll
    for (int t = 0; t < NT; ++t) {
        int col = t * 16 + nn;
        float bv = (MODE != 1) ? bias[col] : 0.f;
#pragma unroll
        for (int r = 0; r < 8; ++r) {
            float v = acc[t][r];
            if (MODE != 1) v = lrelu(v + bv);
            size_t row = (size_t)(r0 + mb + r);
            if (MODE == 0) outH[row * MOUT + col] = (_Float16)v;
            else           outF[row * MOUT + col] = v;
        }
    }
}

// ================= CSR construction (built once, reused by both convs) =========
__global__ __launch_bounds__(256)
void zero_i32(int* __restrict__ p, int n) {
    int i = blockIdx.x * blockDim.x + threadIdx.x;
    if (i < n) p[i] = 0;
}

__global__ __launch_bounds__(256)
void csr_count(const int* __restrict__ dst, int* __restrict__ counts) {
    int e = blockIdx.x * blockDim.x + threadIdx.x;
    if (e < kE) atomicAdd(&counts[dst[e]], 1);
}

// Single-workgroup chunked Hillis-Steele scan: counts[N] -> exclusive starts[N+1]
__global__ __launch_bounds__(1024)
void csr_scan(const int* __restrict__ counts, int* __restrict__ starts) {
    __shared__ int buf[1024];
    __shared__ int running_s;
    int t = threadIdx.x;
    if (t == 0) running_s = 0;
    __syncthreads();
    for (int base = 0; base < kN; base += 1024) {
        int i = base + t;
        int v = (i < kN) ? counts[i] : 0;
        buf[t] = v;
        __syncthreads();
#pragma unroll
        for (int off = 1; off < 1024; off <<= 1) {
            int add = (t >= off) ? buf[t - off] : 0;
            __syncthreads();
            buf[t] += add;
            __syncthreads();
        }
        int run = running_s;        // stable: written after barrier below
        if (i < kN) starts[i] = run + buf[t] - v;   // exclusive prefix
        __syncthreads();
        if (t == 1023) running_s = run + buf[1023];
        __syncthreads();
    }
    if (t == 0) starts[kN] = running_s;
}

// deg[v] = in-degree + 1 (self loop), derived from CSR row extents
__global__ __launch_bounds__(256)
void deg_from_starts(const int* __restrict__ starts, float* __restrict__ dinv,
                     float* __restrict__ dinvr) {
    int i = blockIdx.x * blockDim.x + threadIdx.x;
    if (i >= kN) return;
    float d = (float)(starts[i + 1] - starts[i] + 1);
    dinv[i]  = rsqrtf(d);
    dinvr[i] = 1.0f / d;
}

__global__ __launch_bounds__(256)
void pos_copy(const int* __restrict__ starts, int* __restrict__ pos) {
    int i = blockIdx.x * blockDim.x + threadIdx.x;
    if (i < kN) pos[i] = starts[i];
}

__global__ __launch_bounds__(256)
void csr_fill(const int* __restrict__ src, const int* __restrict__ dst,
              const float* __restrict__ dinv, int* __restrict__ pos,
              int* __restrict__ csr_src, float* __restrict__ csr_w) {
    int e = blockIdx.x * blockDim.x + threadIdx.x;
    if (e >= kE) return;
    int s = src[e], d = dst[e];
    int j = atomicAdd(&pos[d], 1);
    csr_src[j] = s;
    csr_w[j]   = dinv[s];            // dinv[dst] factored out in the gather
}

// ---- fused GCN aggregation (gather, no atomics):
//   h_next[v] = ( sum_in hW[s]*dinv[s] ) * dinv[v] + hW[v]/deg[v] + bias
// one wave per node; lane owns cols l, l+32, ..., l+160; writes f16 for next GEMM
__global__ __launch_bounds__(256)
void gcn_gather(const int* __restrict__ starts, const int* __restrict__ csr_src,
                const float* __restrict__ csr_w, const float* __restrict__ dinv,
                const float* __restrict__ dinvr, const float* __restrict__ hW,
                const float* __restrict__ bias, _Float16* __restrict__ H) {
    int v = blockIdx.x * (blockDim.x >> 5) + (threadIdx.x >> 5);
    if (v >= kN) return;
    int lane = threadIdx.x & 31;
    float a0 = 0.f, a1 = 0.f, a2 = 0.f, a3 = 0.f, a4 = 0.f, a5 = 0.f;
    int jb = starts[v], je = starts[v + 1];
    for (int j = jb; j < je; ++j) {
        int s   = csr_src[j];
        float w = csr_w[j];
        const float* hs = hW + (size_t)s * kH;
        a0 += hs[lane      ] * w;
        a1 += hs[lane +  32] * w;
        a2 += hs[lane +  64] * w;
        a3 += hs[lane +  96] * w;
        a4 += hs[lane + 128] * w;
        a5 += hs[lane + 160] * w;
    }
    float dv = dinv[v], dr = dinvr[v];
    const float* hv = hW + (size_t)v * kH;
    _Float16*    hr = H  + (size_t)v * kH;
    float acc[6] = {a0, a1, a2, a3, a4, a5};
#pragma unroll
    for (int k = 0; k < 6; ++k) {
        int col = lane + 32 * k;
        float val = acc[k] * dv + hv[col] * dr + bias[col];
        hr[col] = (_Float16)val;
    }
}

// ---- final tiny GEMM: out[N,2] = em[N,96] @ W_o2[2,96]^T + b_o2
__global__ __launch_bounds__(256)
void out_gemm(const float* __restrict__ em, const float* __restrict__ W,
              const float* __restrict__ b, float* __restrict__ out) {
    int i = blockIdx.x * blockDim.x + threadIdx.x;
    if (i >= kN) return;
    const float* er = em + (size_t)i * 96;
    float a0 = b[0], a1 = b[1];
#pragma unroll 8
    for (int k = 0; k < 96; ++k) {
        float e = er[k];
        a0 += e * W[k];
        a1 += e * W[96 + k];
    }
    out[2 * i + 0] = a0;
    out[2 * i + 1] = a1;
}

extern "C" void kernel_launch(void* const* d_in, const int* in_sizes, int n_in,
                              void* d_out, int out_size, void* d_ws, size_t ws_size,
                              hipStream_t stream) {
    // ---- inputs (setup_inputs order) ----
    const float* pre_x    = (const float*)d_in[0];
    const float* x        = (const float*)d_in[1];
    const int*   eidx     = (const int*)d_in[2];   // [2, E]
    /* d_in[3] edge_type unused */
    const float* num_prop = (const float*)d_in[4];
    const float* num_cat  = (const float*)d_in[5];
    const float* des      = (const float*)d_in[6];
    const float* tweet    = (const float*)d_in[7];
    const float* W_des = (const float*)d_in[8],  *b_des = (const float*)d_in[9];
    const float* W_twt = (const float*)d_in[10], *b_twt = (const float*)d_in[11];
    const float* W_ttr = (const float*)d_in[12], *b_ttr = (const float*)d_in[13];
    const float* W_prp = (const float*)d_in[14], *b_prp = (const float*)d_in[15];
    const float* W_cat = (const float*)d_in[16], *b_cat = (const float*)d_in[17];
    const float* W_txt = (const float*)d_in[18], *b_txt = (const float*)d_in[19];
    const float* W_in  = (const float*)d_in[20], *b_in  = (const float*)d_in[21];
    const float* W_c1  = (const float*)d_in[22], *b_c1  = (const float*)d_in[23];
    const float* W_c2  = (const float*)d_in[24], *b_c2  = (const float*)d_in[25];
    const float* W_o1  = (const float*)d_in[26], *b_o1  = (const float*)d_in[27];
    const float* W_o2  = (const float*)d_in[28], *b_o2  = (const float*)d_in[29];

    const int* src = eidx;
    const int* dst = eidx + kE;

    // ---- outputs: (out [N,2], em [N,96]) flat concat ----
    float* out = (float*)d_out;
    float* em  = out + (size_t)kN * 2;

    // ---- workspace carve (256B aligned) ----
    char* w = (char*)d_ws;
    auto carve = [&](size_t bytes) -> void* {
        void* p = (void*)w;
        w += (bytes + 255) & ~(size_t)255;
        return p;
    };
    _Float16* hA     = (_Float16*)carve((size_t)kN * kH * sizeof(_Float16));
    _Float16* hB     = (_Float16*)carve((size_t)kN * kH * sizeof(_Float16));
    float*    hW     = (float*)   carve((size_t)kN * kH * sizeof(float));
    int*      counts = (int*)     carve((size_t)kN * sizeof(int));
    int*      starts = (int*)     carve((size_t)(kN + 1) * sizeof(int));
    int*      pos    = (int*)     carve((size_t)kN * sizeof(int));
    int*      csrsrc = (int*)     carve((size_t)kE * sizeof(int));
    float*    csrw   = (float*)   carve((size_t)kE * sizeof(float));
    float*    dinv   = (float*)   carve((size_t)kN * sizeof(float));
    float*    dinvr  = (float*)   carve((size_t)kN * sizeof(float));

    const dim3 blk(256);
    const dim3 gN((kN + 255) / 256);
    const dim3 gTile((kTiles + kWavesPerBlock - 1) / kWavesPerBlock);
    const dim3 gE((kE + 255) / 256);
    const dim3 gNodeWave((kN + kWavesPerBlock - 1) / kWavesPerBlock);

    // Stage 1: branches -> hA [N,192] f16  (concat order: prop, cat, des, text, pre_x, x)
    small_branches<<<gN, blk, 0, stream>>>(num_prop, num_cat, W_prp, b_prp, W_cat, b_cat, hA);
    branch_gemm768<<<gTile, blk, 0, stream>>>(des,   W_des, b_des, hA, 64);
    branch_gemm768<<<gTile, blk, 0, stream>>>(tweet, W_txt, b_txt, hA, 96);
    branch_gemm768<<<gTile, blk, 0, stream>>>(pre_x, W_twt, b_twt, hA, 128);
    branch_gemm768<<<gTile, blk, 0, stream>>>(x,     W_ttr, b_ttr, hA, 160);

    // CSR build (once; reused by both convs)
    zero_i32<<<gN, blk, 0, stream>>>(counts, kN);
    csr_count<<<gE, blk, 0, stream>>>(dst, counts);
    csr_scan<<<dim3(1), dim3(1024), 0, stream>>>(counts, starts);
    deg_from_starts<<<gN, blk, 0, stream>>>(starts, dinv, dinvr);
    pos_copy<<<gN, blk, 0, stream>>>(starts, pos);
    csr_fill<<<gE, blk, 0, stream>>>(src, dst, dinv, pos, csrsrc, csrw);

    // Stage 2: h1 = lrelu(h @ W_in^T + b_in) -> hB f16
    gemm_h<192, 0><<<gTile, blk, 0, stream>>>(hA, W_in, b_in, hB, nullptr);

    // Conv 1: hW = h1 @ W_c1^T ; fused gather epilogue -> hA f16
    gemm_h<192, 1><<<gTile, blk, 0, stream>>>(hB, W_c1, b_c1, nullptr, hW);
    gcn_gather<<<gNodeWave, blk, 0, stream>>>(starts, csrsrc, csrw, dinv, dinvr, hW, b_c1, hA);

    // Conv 2: hW = h2 @ W_c2^T ; fused gather epilogue -> hB f16
    gemm_h<192, 1><<<gTile, blk, 0, stream>>>(hA, W_c2, b_c2, nullptr, hW);
    gcn_gather<<<gNodeWave, blk, 0, stream>>>(starts, csrsrc, csrw, dinv, dinvr, hW, b_c2, hB);

    // Stage 5: em = lrelu(h @ W_o1^T + b_o1) straight into d_out; out = em @ W_o2^T + b_o2
    gemm_h<96, 2><<<gTile, blk, 0, stream>>>(hB, W_o1, b_o1, nullptr, em);
    out_gemm<<<gN, blk, 0, stream>>>(em, W_o2, b_o2, out);
}
